// HyperSpeechMIL_7043746365592
// MI455X (gfx1250) — matrix operand order repaired and checked by reading, not compiled, not run on hardware
//
#include <hip/hip_runtime.h>
#include <hip/hip_bf16.h>

typedef __attribute__((ext_vector_type(16))) _Float16 v16h;
typedef __attribute__((ext_vector_type(8)))  _Float16 v8h;
typedef __attribute__((ext_vector_type(8)))  float    v8f;

#define HSM_B 128
#define HSM_N 2048
#define HSM_D 768
#define HSM_H 64
#define SK_WAVES 2   // waves (16-token tiles) per workgroup in the score kernel

static __device__ inline v16h hsm_cat8(v8h lo, v8h hi) {
  return __builtin_shufflevector(lo, hi, 0,1,2,3,4,5,6,7,8,9,10,11,12,13,14,15);
}

// ---------------------------------------------------------------------------
// Kernel 0: w1 [D][H] f32  ->  w1t [H][D] f16  (so B-fragments are contiguous)
// ---------------------------------------------------------------------------
__global__ void hsm_prep_w1(const float* __restrict__ w1, _Float16* __restrict__ w1t) {
  int i = blockIdx.x * 256 + threadIdx.x;          // over H*D
  if (i < HSM_H * HSM_D) {
    int n = i / HSM_D;
    int k = i - n * HSM_D;
    w1t[i] = (_Float16)w1[(size_t)k * HSM_H + n];
  }
}

// ---------------------------------------------------------------------------
// Kernel 1: per-token LayerNorm -> (xn @ w1) via v_wmma_f32_16x16x32_f16
//           -> tanh -> dot w2 -> scores[B*N]
// One wave handles a 16-token tile; LDS stages the f16 LayerNorm output.
// ---------------------------------------------------------------------------
__global__ __launch_bounds__(32 * SK_WAVES)
void hsm_score_kernel(const float* __restrict__ emb,
                      const float* __restrict__ ln_g,
                      const float* __restrict__ ln_b,
                      const _Float16* __restrict__ w1t,
                      const float* __restrict__ b1,
                      const float* __restrict__ w2,
                      const float* __restrict__ b2,
                      float* __restrict__ scores) {
  __shared__ __align__(16) _Float16 xn[SK_WAVES][16][HSM_D];  // 48 KB

  const int  lane = threadIdx.x & 31;
  const int  wv   = threadIdx.x >> 5;
  const long tile = (long)blockIdx.x * SK_WAVES + wv;
  const long tok0 = tile * 16;
  const float* ebase = emb + tok0 * HSM_D;

  // Per-lane LayerNorm affine params for the 24 columns this lane owns.
  float gg[24], bb[24];
#pragma unroll
  for (int j = 0; j < 24; ++j) { int c = lane + 32 * j; gg[j] = ln_g[c]; bb[j] = ln_b[c]; }

  // ---- cooperative LayerNorm of 16 tokens, f16 result staged to LDS ----
  for (int t = 0; t < 16; ++t) {
    const float* row = ebase + t * HSM_D;
    float v[24];
    float s = 0.f, ss = 0.f;
#pragma unroll
    for (int j = 0; j < 24; ++j) {
      float x = row[lane + 32 * j];     // coalesced: 32 lanes x contiguous
      v[j] = x; s += x; ss += x * x;
    }
#pragma unroll
    for (int off = 16; off >= 1; off >>= 1) {
      s  += __shfl_xor(s,  off, 32);
      ss += __shfl_xor(ss, off, 32);
    }
    const float mean = s * (1.f / HSM_D);
    const float var  = ss * (1.f / HSM_D) - mean * mean;
    const float rstd = rsqrtf(var + 1e-5f);
#pragma unroll
    for (int j = 0; j < 24; ++j)
      xn[wv][t][lane + 32 * j] = (_Float16)((v[j] - mean) * rstd * gg[j] + bb[j]);
  }
  __syncthreads();   // make staged LDS visible across lanes

  // ---- WMMA: 16x64 = (16x768) @ (768x64), K-step 32 ----
  v8f acc[4] = {};
  const int m    = lane & 15;        // A row / C N-col
  const int ksel = lane >> 4;        // 0 or 1 (K half select)
  const _Float16* arow = &xn[wv][m][0];
  const int ncol = lane & 15;

  for (int kk = 0; kk < HSM_D / 32; ++kk) {
    // A fragment (ISA 16-bit 16x32 layout): halves0-7 = K ksel*8+0..7,
    // halves8-15 = K 16+ksel*8+0..7, row = lane&15.
    const int ka = kk * 32 + ksel * 8;
    v16h A = hsm_cat8(*(const v8h*)(arow + ka), *(const v8h*)(arow + ka + 16));

    // B fragment (32x16): halves0-7 = K ksel*8+0..7, halves8-15 = K 16+ksel*8+0..7, N = l&15.
    const int kb = kk * 32 + ksel * 8;
#pragma unroll
    for (int nt = 0; nt < 4; ++nt) {
      const _Float16* bp = w1t + (size_t)(nt * 16 + ncol) * HSM_D + kb;
      v16h Bm = hsm_cat8(*(const v8h*)bp, *(const v8h*)(bp + 16));
      acc[nt] = __builtin_amdgcn_wmma_f32_16x16x32_f16(
          false, A, false, Bm, (short)0, acc[nt], false, false);
    }
  }

  // ---- epilogue: tanh, dot with w2, reduce over N (16 lanes per half) ----
  float part[8] = {0.f, 0.f, 0.f, 0.f, 0.f, 0.f, 0.f, 0.f};
#pragma unroll
  for (int nt = 0; nt < 4; ++nt) {
    const float b1v = b1[nt * 16 + ncol];
    const float w2v = w2[nt * 16 + ncol];
#pragma unroll
    for (int v = 0; v < 8; ++v)
      part[v] += tanhf(acc[nt][v] + b1v) * w2v;   // C/D: M = v + 8*ksel, N = ncol
  }
  const float b2v = b2[0];
#pragma unroll
  for (int v = 0; v < 8; ++v) {
    float sv = part[v];
    sv += __shfl_xor(sv, 1, 32);
    sv += __shfl_xor(sv, 2, 32);
    sv += __shfl_xor(sv, 4, 32);
    sv += __shfl_xor(sv, 8, 32);
    if ((lane & 15) == 0)
      scores[tok0 + v + 8 * ksel] = sv + b2v;
  }
}

// ---------------------------------------------------------------------------
// Block reduction helper (256 threads)
// ---------------------------------------------------------------------------
static __device__ inline float hsm_br(float v, float* buf, bool ismax) {
  const int t = threadIdx.x;
  buf[t] = v; __syncthreads();
  for (int s = 128; s > 0; s >>= 1) {
    if (t < s) buf[t] = ismax ? fmaxf(buf[t], buf[t + s]) : (buf[t] + buf[t + s]);
    __syncthreads();
  }
  float r = buf[0]; __syncthreads();
  return r;
}

// ---------------------------------------------------------------------------
// Kernel 2: per-batch masked softmax over scores + logit statistics
// ---------------------------------------------------------------------------
__global__ __launch_bounds__(256)
void hsm_softmax_stats_kernel(const float* __restrict__ logits,
                              const unsigned char* __restrict__ mask,
                              const float* __restrict__ scores,
                              float* __restrict__ weights,
                              float* __restrict__ stats) {
  __shared__ float red[256];
  __shared__ float tops[256][3];
  const int b = blockIdx.x, t = threadIdx.x;
  const float* sc = scores + (size_t)b * HSM_N;
  const float* lg = logits + (size_t)b * HSM_N;
  const unsigned char* mk = mask + (size_t)b * HSM_N;
  float* wrow = weights + (size_t)b * HSM_N;

  float smax = -3.0e38f, lsum = 0.f, lsq = 0.f, lmax = -3.0e38f, cntf = 0.f;
  float t0 = -3.0e38f, t1 = -3.0e38f, t2 = -3.0e38f;
  for (int n = t; n < HSM_N; n += 256) {
    if (mk[n]) {
      float s = sc[n]; smax = fmaxf(smax, s);
      float x = lg[n];
      lsum += x; lsq += x * x; cntf += 1.f; lmax = fmaxf(lmax, x);
      if (x > t0)      { t2 = t1; t1 = t0; t0 = x; }
      else if (x > t1) { t2 = t1; t1 = x; }
      else if (x > t2) { t2 = x; }
    }
  }
  const float smaxT = hsm_br(smax, red, true);
  const float lsumT = hsm_br(lsum, red, false);
  const float lsqT  = hsm_br(lsq,  red, false);
  const float lmaxT = hsm_br(lmax, red, true);
  const float cntT  = hsm_br(cntf, red, false);
  tops[t][0] = t0; tops[t][1] = t1; tops[t][2] = t2;
  __syncthreads();

  float sume = 0.f;
  for (int n = t; n < HSM_N; n += 256) {
    float w = mk[n] ? __expf(sc[n] - smaxT) : 0.f;
    wrow[n] = w;
    sume += w;
  }
  const float inv = 1.f / hsm_br(sume, red, false);
  for (int n = t; n < HSM_N; n += 256) wrow[n] *= inv;

  if (t == 0) {
    float a0 = -3.0e38f, a1 = -3.0e38f, a2 = -3.0e38f;
    for (int i = 0; i < 256; ++i)
      for (int j = 0; j < 3; ++j) {
        float x = tops[i][j];
        if (x > a0)      { a2 = a1; a1 = a0; a0 = x; }
        else if (x > a1) { a2 = a1; a1 = x; }
        else if (x > a2) { a2 = x; }
      }
    const float cnt  = fmaxf(cntT, 1.f);
    const float mean = lsumT / cnt;
    const float var  = lsqT / cnt - mean * mean;
    stats[b * 4 + 0] = mean;
    stats[b * 4 + 1] = sqrtf(fmaxf(var, 0.f) + 1e-12f);
    stats[b * 4 + 2] = lmaxT;
    stats[b * 4 + 3] = (a0 + a1 + a2) * (1.f / 3.f);
  }
}

// ---------------------------------------------------------------------------
// Kernel 3a: partial weighted pool over 128-token chunks (grid = B x 16)
// ---------------------------------------------------------------------------
__global__ __launch_bounds__(256)
void hsm_pool_partial_kernel(const float* __restrict__ emb,
                             const float* __restrict__ weights,
                             float* __restrict__ poolp) {
  const int b = blockIdx.x, chunk = blockIdx.y, t = threadIdx.x;
  __shared__ float wsh[128];
  if (t < 128) wsh[t] = weights[(size_t)b * HSM_N + chunk * 128 + t];
  __syncthreads();

  const float* eb = emb + ((size_t)b * HSM_N + (size_t)chunk * 128) * HSM_D;
  const int d0 = t, d1 = t + 256, d2 = t + 512;
  float a0 = 0.f, a1 = 0.f, a2 = 0.f;
  for (int n = 0; n < 128; ++n) {
    const float* row = eb + n * HSM_D;
    __builtin_prefetch(row + HSM_D, 0, 0);   // global_prefetch_b8 for next row
    const float w = wsh[n];
    a0 += w * row[d0];
    a1 += w * row[d1];
    a2 += w * row[d2];
  }
  float* outp = poolp + ((size_t)b * 16 + chunk) * HSM_D;
  outp[d0] = a0; outp[d1] = a1; outp[d2] = a2;
}

// ---------------------------------------------------------------------------
// Kernel 3b: reduce partials, concat stats, subject head -> out[B]
// ---------------------------------------------------------------------------
__global__ __launch_bounds__(256)
void hsm_head_kernel(const float* __restrict__ poolp,
                     const float* __restrict__ stats,
                     const float* __restrict__ wh1,
                     const float* __restrict__ bh1,
                     const float* __restrict__ wh2,
                     const float* __restrict__ bh2,
                     float* __restrict__ out) {
  const int b = blockIdx.x, t = threadIdx.x;
  __shared__ float feat[HSM_D + 4];
  __shared__ float red[64];
  const float* pp = poolp + (size_t)b * 16 * HSM_D;
#pragma unroll
  for (int i = 0; i < 3; ++i) {
    const int d = t + i * 256;
    float s = 0.f;
    for (int c = 0; c < 16; ++c) s += pp[c * HSM_D + d];
    feat[d] = s;
  }
  if (t < 4) feat[HSM_D + t] = stats[b * 4 + t];
  __syncthreads();

  if (t < HSM_H) {
    float s = bh1[t];
    for (int d = 0; d < HSM_D + 4; ++d) s += feat[d] * wh1[(size_t)d * HSM_H + t];
    red[t] = fmaxf(s, 0.f) * wh2[t];
  }
  __syncthreads();
  if (t == 0) {
    float s = 0.f;
    for (int h = 0; h < HSM_H; ++h) s += red[h];
    out[b] = s + bh2[0];
  }
}

// ---------------------------------------------------------------------------
// Launch
// ---------------------------------------------------------------------------
extern "C" void kernel_launch(void* const* d_in, const int* in_sizes, int n_in,
                              void* d_out, int out_size, void* d_ws, size_t ws_size,
                              hipStream_t stream) {
  (void)in_sizes; (void)n_in; (void)out_size; (void)ws_size;

  const float*         emb    = (const float*)d_in[0];
  const float*         logits = (const float*)d_in[1];
  const unsigned char* mask   = (const unsigned char*)d_in[2];
  const float*         ln_g   = (const float*)d_in[3];
  const float*         ln_b   = (const float*)d_in[4];
  const float*         w1     = (const float*)d_in[5];
  const float*         b1     = (const float*)d_in[6];
  const float*         w2     = (const float*)d_in[7];
  const float*         b2     = (const float*)d_in[8];
  const float*         wh1    = (const float*)d_in[9];
  const float*         bh1    = (const float*)d_in[10];
  const float*         wh2    = (const float*)d_in[11];
  const float*         bh2    = (const float*)d_in[12];
  float*               out    = (float*)d_out;

  // Workspace layout (256B-aligned regions)
  char* ws = (char*)d_ws;
  const size_t OFF_W1T     = 0;                                   // 98304 B
  const size_t OFF_SCORES  = OFF_W1T    + ((size_t)HSM_H * HSM_D * 2 + 255 & ~(size_t)255);
  const size_t OFF_WEIGHTS = OFF_SCORES + ((size_t)HSM_B * HSM_N * 4);
  const size_t OFF_STATS   = OFF_WEIGHTS+ ((size_t)HSM_B * HSM_N * 4);
  const size_t OFF_POOLP   = OFF_STATS  + ((size_t)HSM_B * 4 * 4 + 255 & ~(size_t)255);

  _Float16* w1t     = (_Float16*)(ws + OFF_W1T);
  float*    scores  = (float*)(ws + OFF_SCORES);
  float*    weights = (float*)(ws + OFF_WEIGHTS);
  float*    stats   = (float*)(ws + OFF_STATS);
  float*    poolp   = (float*)(ws + OFF_POOLP);

  hsm_prep_w1<<<(HSM_H * HSM_D + 255) / 256, 256, 0, stream>>>(w1, w1t);

  const int tiles = (HSM_B * HSM_N) / 16;             // 16384
  hsm_score_kernel<<<tiles / SK_WAVES, 32 * SK_WAVES, 0, stream>>>(
      emb, ln_g, ln_b, w1t, b1, w2, b2, scores);

  hsm_softmax_stats_kernel<<<HSM_B, 256, 0, stream>>>(
      logits, mask, scores, weights, stats);

  hsm_pool_partial_kernel<<<dim3(HSM_B, 16), 256, 0, stream>>>(
      emb, weights, poolp);

  hsm_head_kernel<<<HSM_B, 256, 0, stream>>>(
      poolp, stats, wh1, bh1, wh2, bh2, out);
}
